// _DecoderLayer_4312147165305
// MI455X (gfx1250) — matrix operand order repaired and checked by reading, not compiled, not run on hardware
//
#include <hip/hip_runtime.h>
#include <math.h>

typedef __attribute__((ext_vector_type(16))) _Float16 v16h;
typedef __attribute__((ext_vector_type(8)))  _Float16 v8h;
typedef __attribute__((ext_vector_type(4)))  _Float16 v4h;
typedef __attribute__((ext_vector_type(8)))  float    v8f;
typedef __attribute__((ext_vector_type(4)))  float    v4f;

static constexpr int Bb  = 4;
static constexpr int Kq  = 512;
static constexpr int Sv  = 8192;
static constexpr int Dm  = 512;
static constexpr int Hh  = 8;
static constexpr int HID = 2048;
static constexpr int HD  = 64;

// ---------------------------------------------------------------- fragments
// A fragment, f16 source: per lane two contiguous 16B chunks:
//   elems 0..7  <- p[8*hi .. 8*hi+7]       (K = 8hi..8hi+7)
//   elems 8..15 <- p[16+8*hi .. 16+8*hi+7] (K = 16+8hi..23+8hi)
__device__ __forceinline__ v16h load_a_f16(const _Float16* p, int hi) {
  v8h lo = *(const v8h*)(p + 8 * hi);
  v8h hh = *(const v8h*)(p + 16 + 8 * hi);
  return __builtin_shufflevector(lo, hh, 0, 1, 2, 3, 4, 5, 6, 7,
                                 8, 9, 10, 11, 12, 13, 14, 15);
}

// B fragment: lane = column, elems 0..7 -> K = 8hi..8hi+7, elems 8..15 -> K = 16+8hi..23+8hi
__device__ __forceinline__ v16h load_b_f16(const _Float16* p, int hi) {
  v8h lo = *(const v8h*)(p + 8 * hi);
  v8h hh = *(const v8h*)(p + 16 + 8 * hi);
  return __builtin_shufflevector(lo, hh, 0, 1, 2, 3, 4, 5, 6, 7,
                                 8, 9, 10, 11, 12, 13, 14, 15);
}

__device__ __forceinline__ v16h cvt_pack(v4f a0, v4f a1, v4f b0, v4f b1) {
  v4h c0 = __builtin_convertvector(a0, v4h);
  v4h c1 = __builtin_convertvector(a1, v4h);
  v4h c2 = __builtin_convertvector(b0, v4h);
  v4h c3 = __builtin_convertvector(b1, v4h);
  v8h lo = __builtin_shufflevector(c0, c1, 0, 1, 2, 3, 4, 5, 6, 7);
  v8h hh = __builtin_shufflevector(c2, c3, 0, 1, 2, 3, 4, 5, 6, 7);
  return __builtin_shufflevector(lo, hh, 0, 1, 2, 3, 4, 5, 6, 7,
                                 8, 9, 10, 11, 12, 13, 14, 15);
}

__device__ __forceinline__ v16h load_a_f32(const float* p, int hi) {
  const v4f* q0 = (const v4f*)(p + 8 * hi);
  const v4f* q1 = (const v4f*)(p + 16 + 8 * hi);
  return cvt_pack(q0[0], q0[1], q1[0], q1[1]);
}

__device__ __forceinline__ v16h load_a_f32_add(const float* p, const float* p2, int hi) {
  const v4f* q0 = (const v4f*)(p + 8 * hi);
  const v4f* q1 = (const v4f*)(p + 16 + 8 * hi);
  const v4f* r0 = (const v4f*)(p2 + 8 * hi);
  const v4f* r1 = (const v4f*)(p2 + 16 + 8 * hi);
  return cvt_pack(q0[0] + r0[0], q0[1] + r0[1], q1[0] + r1[0], q1[1] + r1[1]);
}

__device__ __forceinline__ v8f wmma32(v16h a, v16h b, v8f c) {
  return __builtin_amdgcn_wmma_f32_16x16x32_f16(false, a, false, b, (short)0, c,
                                                false, false);
}

// ---------------------------------------------------------------- GEMM
// C[M,N] = A[M,Kd] * Wt^T + bias, Wt stored [N,Kd] f16.
// Block: 256 thr = 8 waves stacked in M; block tile 128x64.
// Wave: 16 rows x 64 cols (4 WMMA tiles, one shared A fragment per K-chunk).
// B chunk (64 cols x 32 K) staged in LDS, double-buffered.
enum { EPI_F16 = 0, EPI_GATE = 1, EPI_RES_F32 = 2, EPI_GELU = 3, EPI_F32 = 4,
       EPI_F16T = 5 /* write V transposed per head: vT[B,H,HD,S] */ };

template <bool AF16, bool ADDA, int EPI>
__global__ __launch_bounds__(256) void gemm_wmma(
    const void* A_, const float* Aadd, const _Float16* Wt, const float* bias,
    const float* gate, const float* resid, void* C_, int M, int N, int Kd) {
  __shared__ _Float16 bsh[2][64 * 32];   // [buf][col][k] 4KB each
  __shared__ float    csh[8][16 * 65];   // per-wave epilogue transpose

  const int t = threadIdx.x;
  const int lane = t & 31;
  const int lr = lane & 15, hi = lane >> 4;
  const int w = t >> 5;
  const int rowBase = blockIdx.y * 128 + w * 16;
  const int colBase = blockIdx.x * 64;
  const int arow = rowBase + lr;

  const float* Af = (const float*)A_;
  const _Float16* Ah = (const _Float16*)A_;

  // staging role: thread t copies 8 halves of Wt row (colBase + t/4)
  const int sn = t >> 2, skp = (t & 3) * 8;
  const _Float16* ssrc = Wt + (size_t)(colBase + sn) * Kd + skp;

  *(v8h*)(&bsh[0][sn * 32 + skp]) = *(const v8h*)(ssrc);
  __syncthreads();

  v8f acc0 = {}, acc1 = {}, acc2 = {}, acc3 = {};
  int buf = 0;
  for (int ck = 0; ck < Kd; ck += 32) {
    if (ck + 32 < Kd)
      *(v8h*)(&bsh[buf ^ 1][sn * 32 + skp]) = *(const v8h*)(ssrc + ck + 32);

    v16h a;
    if (AF16)
      a = load_a_f16(Ah + (size_t)arow * Kd + ck, hi);
    else if (ADDA)
      a = load_a_f32_add(Af + (size_t)arow * Kd + ck,
                         Aadd + (size_t)arow * Kd + ck, hi);
    else
      a = load_a_f32(Af + (size_t)arow * Kd + ck, hi);

    const _Float16* bb = &bsh[buf][16 * hi];
    v16h b0 = load_b_f16(&bsh[buf][( 0 + lr) * 32], hi);
    v16h b1 = load_b_f16(&bsh[buf][(16 + lr) * 32], hi);
    v16h b2 = load_b_f16(&bsh[buf][(32 + lr) * 32], hi);
    v16h b3 = load_b_f16(&bsh[buf][(48 + lr) * 32], hi);
    (void)bb;
    acc0 = wmma32(a, b0, acc0);
    acc1 = wmma32(a, b1, acc1);
    acc2 = wmma32(a, b2, acc2);
    acc3 = wmma32(a, b3, acc3);
    __syncthreads();
    buf ^= 1;
  }

  if (EPI == EPI_F16T) {
    // per lane: col n fixed per tile, rows rowBase+8*hi+r consecutive -> s
    const int m0 = rowBase + 8 * hi;
    const int bidx = m0 >> 13;              // / Sv
    const int s = m0 & (Sv - 1);
    v8f* accs[4] = {&acc0, &acc1, &acc2, &acc3};
#pragma unroll
    for (int c = 0; c < 4; ++c) {
      const int n = colBase + c * 16 + lr;
      const int h = n >> 6, hd = n & (HD - 1);
      const float bv = bias[n];
      v8h o;
#pragma unroll
      for (int r = 0; r < 8; ++r) o[r] = (_Float16)((*accs[c])[r] + bv);
      *(v8h*)((_Float16*)C_ + ((size_t)(bidx * Hh + h) * HD + hd) * Sv + s) = o;
    }
    return;
  }

  // LDS transpose of the 16x64 wave strip: lane -> row lr, cols 32*hi..+31
  float* tw = csh[w];
  v8f* accs[4] = {&acc0, &acc1, &acc2, &acc3};
#pragma unroll
  for (int c = 0; c < 4; ++c)
#pragma unroll
    for (int r = 0; r < 8; ++r)
      tw[(r + 8 * hi) * 65 + c * 16 + lr] = (*accs[c])[r];
  __syncthreads();

  const int rowm = rowBase + lr;
  const int n0 = colBase + 32 * hi;
  float vals[32];
#pragma unroll
  for (int j = 0; j < 32; ++j) vals[j] = tw[lr * 65 + 32 * hi + j];
  const v4f* bp = (const v4f*)(bias + n0);
#pragma unroll
  for (int q = 0; q < 8; ++q) {
    v4f bb = bp[q];
#pragma unroll
    for (int j = 0; j < 4; ++j) vals[q * 4 + j] += bb[j];
  }
  const size_t idx0 = (size_t)rowm * N + n0;

  if (EPI == EPI_F16 || EPI == EPI_GATE || EPI == EPI_GELU) {
    float gm = 1.0f;
    if (EPI == EPI_GATE) gm = gate[rowm] * 0.125f;  // gate * hd^-0.5
#pragma unroll
    for (int q = 0; q < 4; ++q) {
      v8h o;
#pragma unroll
      for (int j = 0; j < 8; ++j) {
        float y = vals[q * 8 + j];
        if (EPI == EPI_GATE) y *= gm;
        if (EPI == EPI_GELU) y = 0.5f * y * (1.0f + erff(y * 0.70710678118654752f));
        o[j] = (_Float16)y;
      }
      ((v8h*)((_Float16*)C_ + idx0))[q] = o;
    }
  } else {  // EPI_RES_F32 / EPI_F32
#pragma unroll
    for (int q = 0; q < 8; ++q) {
      v4f o;
#pragma unroll
      for (int j = 0; j < 4; ++j) o[j] = vals[q * 4 + j];
      if (EPI == EPI_RES_F32) o += ((const v4f*)(resid + idx0))[q];
      ((v4f*)((float*)C_ + idx0))[q] = o;
    }
  }
}

// ---------------------------------------------------------------- attention
// Per wave: 16 queries of one (b,h); scores^T = K_tile x q^T; online softmax;
// out^T = V^T x P^T accumulated over S.  vT layout: [B,H,HD,S] f16.
__global__ __launch_bounds__(128) void attn_kernel(const _Float16* qh,
                                                   const _Float16* kh,
                                                   const _Float16* vT,
                                                   const float* bias,
                                                   float* out) {
  const int lane = threadIdx.x & 31;
  const int lr = lane & 15, hi = lane >> 4;
  const int w = threadIdx.x >> 5;
  const int b = blockIdx.z, h = blockIdx.y;
  const int q0 = blockIdx.x * 64 + w * 16;

  const _Float16* qbase = qh + (size_t)b * Kq * Dm + h * HD;
  const _Float16* kbase = kh + (size_t)b * Sv * Dm + h * HD;
  const _Float16* vtb = vT + (size_t)(b * Hh + h) * HD * Sv;
  const float* brow = bias + ((size_t)b * Kq + q0 + lr) * Sv;

  const _Float16* qrow = qbase + (size_t)(q0 + lr) * Dm;
  const v16h qb0 = load_b_f16(qrow + 0, hi);
  const v16h qb1 = load_b_f16(qrow + 32, hi);

  v8f acc0 = {}, acc1 = {}, acc2 = {}, acc3 = {};
  float mrun = -1e30f, lrun = 0.0f;

  for (int s0 = 0; s0 < Sv; s0 += 32) {
    if (s0 + 32 < Sv) {
      __builtin_prefetch(kbase + (size_t)(s0 + 32 + lr) * Dm, 0, 1);
      __builtin_prefetch(vtb + (size_t)(lane * 2) * Sv + s0 + 32, 0, 1);
    }
    const _Float16* krow0 = kbase + (size_t)(s0 + lr) * Dm;
    const _Float16* krow1 = kbase + (size_t)(s0 + 16 + lr) * Dm;
    v8f c0 = {}, c1 = {};
    c0 = wmma32(load_a_f16(krow0 + 0, hi), qb0, c0);
    c0 = wmma32(load_a_f16(krow0 + 32, hi), qb1, c0);
    c1 = wmma32(load_a_f16(krow1 + 0, hi), qb0, c1);
    c1 = wmma32(load_a_f16(krow1 + 32, hi), qb1, c1);

    const v4f* bp0 = (const v4f*)(brow + s0 + 8 * hi);
    const v4f* bp1 = (const v4f*)(brow + s0 + 16 + 8 * hi);
    v4f ba0 = bp0[0], ba1 = bp0[1], bc0 = bp1[0], bc1 = bp1[1];

    float p0[8], p1[8];
    float cm = -1e30f;
#pragma unroll
    for (int r = 0; r < 8; ++r) {
      p0[r] = c0[r] + ((r < 4) ? ba0[r] : ba1[r - 4]);
      p1[r] = c1[r] + ((r < 4) ? bc0[r] : bc1[r - 4]);
      cm = fmaxf(cm, fmaxf(p0[r], p1[r]));
    }
    cm = fmaxf(cm, __shfl_xor(cm, 16, 32));
    float mnew = fmaxf(mrun, cm);
    float alpha = __expf(mrun - mnew);
    float ls = 0.0f;
#pragma unroll
    for (int r = 0; r < 8; ++r) {
      p0[r] = __expf(p0[r] - mnew);
      p1[r] = __expf(p1[r] - mnew);
      ls += p0[r] + p1[r];
    }
    ls += __shfl_xor(ls, 16, 32);
    lrun = lrun * alpha + ls;
    mrun = mnew;
#pragma unroll
    for (int r = 0; r < 8; ++r) {
      acc0[r] *= alpha; acc1[r] *= alpha; acc2[r] *= alpha; acc3[r] *= alpha;
    }
    // P^T B-fragment: lane owns query col; K(s) = 8*hi + e (e<8), 16 + 8*hi + (e-8) (e>=8)
    v16h pb;
#pragma unroll
    for (int r = 0; r < 8; ++r) {
      pb[r] = (_Float16)p0[r]; pb[r + 8] = (_Float16)p1[r];
    }
    // out^T: A = vT rows (hd), contiguous along s
    acc0 = wmma32(load_a_f16(vtb + (size_t)( 0 + lr) * Sv + s0, hi), pb, acc0);
    acc1 = wmma32(load_a_f16(vtb + (size_t)(16 + lr) * Sv + s0, hi), pb, acc1);
    acc2 = wmma32(load_a_f16(vtb + (size_t)(32 + lr) * Sv + s0, hi), pb, acc2);
    acc3 = wmma32(load_a_f16(vtb + (size_t)(48 + lr) * Sv + s0, hi), pb, acc3);
  }
  const float inv = 1.0f / lrun;
  float* orow = out + ((size_t)b * Kq + q0 + lr) * Dm + h * HD + 8 * hi;
  v4f s0v, s1v;
#pragma unroll
  for (int j = 0; j < 4; ++j) { s0v[j] = acc0[j] * inv; s1v[j] = acc0[4 + j] * inv; }
  ((v4f*)(orow + 0))[0] = s0v; ((v4f*)(orow + 0))[1] = s1v;
#pragma unroll
  for (int j = 0; j < 4; ++j) { s0v[j] = acc1[j] * inv; s1v[j] = acc1[4 + j] * inv; }
  ((v4f*)(orow + 16))[0] = s0v; ((v4f*)(orow + 16))[1] = s1v;
#pragma unroll
  for (int j = 0; j < 4; ++j) { s0v[j] = acc2[j] * inv; s1v[j] = acc2[4 + j] * inv; }
  ((v4f*)(orow + 32))[0] = s0v; ((v4f*)(orow + 32))[1] = s1v;
#pragma unroll
  for (int j = 0; j < 4; ++j) { s0v[j] = acc3[j] * inv; s1v[j] = acc3[4 + j] * inv; }
  ((v4f*)(orow + 48))[0] = s0v; ((v4f*)(orow + 48))[1] = s1v;
}

// ---------------------------------------------------------------- layernorm
__global__ __launch_bounds__(256) void ln_kernel(const float* x, const float* add,
                                                 const float* g, const float* be,
                                                 float* out32, _Float16* out16) {
  __shared__ float red0[8], red1[8];
  const int row = blockIdx.x;
  const int t = threadIdx.x;
  const float* xr = x + (size_t)row * Dm;
  float v0 = xr[t], v1 = xr[t + 256];
  if (add) { v0 += add[(size_t)row * Dm + t]; v1 += add[(size_t)row * Dm + t + 256]; }
  float s = v0 + v1, ss = v0 * v0 + v1 * v1;
#pragma unroll
  for (int off = 16; off > 0; off >>= 1) {
    s += __shfl_xor(s, off, 32);
    ss += __shfl_xor(ss, off, 32);
  }
  if ((t & 31) == 0) { red0[t >> 5] = s; red1[t >> 5] = ss; }
  __syncthreads();
  float S = 0.0f, SS = 0.0f;
#pragma unroll
  for (int i = 0; i < 8; ++i) { S += red0[i]; SS += red1[i]; }
  float mean = S * (1.0f / Dm);
  float var = SS * (1.0f / Dm) - mean * mean;
  float rs = rsqrtf(var + 1e-5f);
  float y0 = (v0 - mean) * rs * g[t] + be[t];
  float y1 = (v1 - mean) * rs * g[t + 256] + be[t + 256];
  out32[(size_t)row * Dm + t] = y0;
  out32[(size_t)row * Dm + t + 256] = y1;
  if (out16) {
    out16[(size_t)row * Dm + t] = (_Float16)y0;
    out16[(size_t)row * Dm + t + 256] = (_Float16)y1;
  }
}

// ---------------------------------------------------------------- transpose
// W [Kd,N] f32 -> Wt [N,Kd] f16 (output-coalesced)
__global__ void transpose_f16(const float* W, _Float16* Wt, int Kd, int N) {
  size_t i = (size_t)blockIdx.x * 256 + threadIdx.x;
  if (i >= (size_t)Kd * N) return;
  int n = (int)(i / Kd), k = (int)(i % Kd);
  Wt[i] = (_Float16)W[(size_t)k * N + n];
}

// ---------------------------------------------------------------- host
extern "C" void kernel_launch(void* const* d_in, const int* in_sizes, int n_in,
                              void* d_out, int out_size, void* d_ws, size_t ws_size,
                              hipStream_t stream) {
  const float* query = (const float*)d_in[0];
  const float* value = (const float*)d_in[1];
  const float* pos   = (const float*)d_in[2];
  const float* abias = (const float*)d_in[3];
  const float* gate  = (const float*)d_in[4];
  const float* Wq = (const float*)d_in[5];  const float* bq = (const float*)d_in[6];
  const float* Wk = (const float*)d_in[7];  const float* bk = (const float*)d_in[8];
  const float* Wv = (const float*)d_in[9];  const float* bv = (const float*)d_in[10];
  const float* Wo = (const float*)d_in[11]; const float* bo = (const float*)d_in[12];
  const float* g1 = (const float*)d_in[13]; const float* be1 = (const float*)d_in[14];
  const float* g2 = (const float*)d_in[15]; const float* be2 = (const float*)d_in[16];
  const float* W1 = (const float*)d_in[17]; const float* b1 = (const float*)d_in[18];
  const float* W2 = (const float*)d_in[19]; const float* b2 = (const float*)d_in[20];

  char* ws = (char*)d_ws;
  size_t off = 0;
  auto alloc = [&](size_t bytes) -> void* {
    void* p = ws + off;
    off = (off + bytes + 255) & ~(size_t)255;
    return p;
  };
  _Float16* qh   = (_Float16*)alloc((size_t)2 * Bb * Kq * Dm);
  _Float16* kh   = (_Float16*)alloc((size_t)2 * Bb * Sv * Dm);
  _Float16* vT   = (_Float16*)alloc((size_t)2 * Bb * Sv * Dm);  // [B,H,HD,S]
  float*    attn = (float*)alloc((size_t)4 * Bb * Kq * Dm);
  float*    ybuf = (float*)alloc((size_t)4 * Bb * Kq * Dm);
  float*    x1   = (float*)alloc((size_t)4 * Bb * Kq * Dm);
  _Float16* x1h  = (_Float16*)alloc((size_t)2 * Bb * Kq * Dm);
  _Float16* hid  = (_Float16*)alloc((size_t)2 * Bb * Kq * HID);
  float*    ffn2 = (float*)alloc((size_t)4 * Bb * Kq * Dm);
  _Float16* Wqt = (_Float16*)alloc((size_t)2 * Dm * Dm);
  _Float16* Wkt = (_Float16*)alloc((size_t)2 * Dm * Dm);
  _Float16* Wvt = (_Float16*)alloc((size_t)2 * Dm * Dm);
  _Float16* Wot = (_Float16*)alloc((size_t)2 * Dm * Dm);
  _Float16* W1t = (_Float16*)alloc((size_t)2 * Dm * HID);
  _Float16* W2t = (_Float16*)alloc((size_t)2 * HID * Dm);

  // weight transposes to f16 [N,Kd]
  transpose_f16<<<(Dm * Dm + 255) / 256, 256, 0, stream>>>(Wq, Wqt, Dm, Dm);
  transpose_f16<<<(Dm * Dm + 255) / 256, 256, 0, stream>>>(Wk, Wkt, Dm, Dm);
  transpose_f16<<<(Dm * Dm + 255) / 256, 256, 0, stream>>>(Wv, Wvt, Dm, Dm);
  transpose_f16<<<(Dm * Dm + 255) / 256, 256, 0, stream>>>(Wo, Wot, Dm, Dm);
  transpose_f16<<<(Dm * HID + 255) / 256, 256, 0, stream>>>(W1, W1t, Dm, HID);
  transpose_f16<<<(HID * Dm + 255) / 256, 256, 0, stream>>>(W2, W2t, HID, Dm);

  dim3 blk(256);
  // q = (query@Wq + bq) * gate * hd^-0.5  -> f16
  gemm_wmma<false, false, EPI_GATE><<<dim3(Dm / 64, (Bb * Kq) / 128), blk, 0, stream>>>(
      query, nullptr, Wqt, bq, gate, nullptr, qh, Bb * Kq, Dm, Dm);
  // k = (value+pos)@Wk + bk -> f16
  gemm_wmma<false, true, EPI_F16><<<dim3(Dm / 64, (Bb * Sv) / 128), blk, 0, stream>>>(
      value, pos, Wkt, bk, nullptr, nullptr, kh, Bb * Sv, Dm, Dm);
  // v = value@Wv + bv -> f16, written transposed per head into vT[B,H,HD,S]
  gemm_wmma<false, false, EPI_F16T><<<dim3(Dm / 64, (Bb * Sv) / 128), blk, 0, stream>>>(
      value, nullptr, Wvt, bv, nullptr, nullptr, vT, Bb * Sv, Dm, Dm);
  // flash attention -> attn f32 [B,K,D]
  attn_kernel<<<dim3(Kq / 64, Hh, Bb), dim3(128), 0, stream>>>(qh, kh, vT, abias, attn);
  // ybuf = attn@Wo + bo + query
  gemm_wmma<false, false, EPI_RES_F32><<<dim3(Dm / 64, (Bb * Kq) / 128), blk, 0, stream>>>(
      attn, nullptr, Wot, bo, nullptr, query, ybuf, Bb * Kq, Dm, Dm);
  // x1 = LN1(ybuf)
  ln_kernel<<<Bb * Kq, blk, 0, stream>>>(ybuf, nullptr, g1, be1, x1, x1h);
  // hid = gelu(x1@W1 + b1) -> f16
  gemm_wmma<true, false, EPI_GELU><<<dim3(HID / 64, (Bb * Kq) / 128), blk, 0, stream>>>(
      x1h, nullptr, W1t, b1, nullptr, nullptr, hid, Bb * Kq, HID, Dm);
  // ffn2 = hid@W2 + b2 -> f32
  gemm_wmma<true, false, EPI_F32><<<dim3(Dm / 64, (Bb * Kq) / 128), blk, 0, stream>>>(
      hid, nullptr, W2t, b2, nullptr, nullptr, ffn2, Bb * Kq, Dm, HID);
  // out = LN2(x1 + ffn2)
  ln_kernel<<<Bb * Kq, blk, 0, stream>>>(x1, ffn2, g2, be2, (float*)d_out, nullptr);

  (void)in_sizes; (void)n_in; (void)out_size; (void)ws_size;
}
